// TransformerEncoder_22385369546769
// MI455X (gfx1250) — matrix operand order repaired and checked by reading, not compiled, not run on hardware
//
#include <hip/hip_runtime.h>
#include <hip/hip_bf16.h>
#include <cmath>

// ---------------------------------------------------------------------------
// Transformer encoder layer on gfx1250 (MI455X), wave32 + WMMA f16->f32.
// All GEMMs (QKV, O, MLP1, MLP2) and both attention matmuls use
// v_wmma_f32_16x16x32_f16. LayerNorm/softmax/residuals stay in f32.
// GEMM: 128x128 block tile, K=64 LDS stage, register-prefetch double buffer.
// ---------------------------------------------------------------------------

typedef _Float16 f16;
typedef __attribute__((ext_vector_type(16))) _Float16 v16h;
typedef __attribute__((ext_vector_type(8)))  _Float16 v8h;
typedef __attribute__((ext_vector_type(8)))  float    v8f;

#define FEATS   1024
#define NSEQ    1024
#define NBATCH  8
#define NHEAD   16
#define DHEAD   64
#define MLPH    4096
#define NTOK    (NBATCH * NSEQ)   // 8192 rows

// ---------------- WMMA fragment helpers (ISA 7.12.2 layouts) ----------------

// A-matrix 16x32 f16: lanes 0-15 row M=lane, VGPR0-3 = K0..7, VGPR4-7 = K16..23
//                     lanes 16-31 row M=lane-16, K8..15 and K24..31.
__device__ __forceinline__ v16h load_fragA(const f16* tile, int row0, int col0, int ldt) {
  const int lane = threadIdx.x & 31;
  const f16* p = tile + (size_t)(row0 + (lane & 15)) * ldt + col0 + ((lane >> 4) << 3);
  v8h lo = *(const v8h*)p;          // 8 contiguous f16
  v8h hi = *(const v8h*)(p + 16);   // +32 bytes
  return __builtin_shufflevector(lo, hi, 0,1,2,3,4,5,6,7,8,9,10,11,12,13,14,15);
}

// B-matrix 32x16 f16 loaded from a [N][K]-row-major tile:
// lanes 0-15 col N=lane hold K0..7 and K16..23; lanes 16-31 hold K8..15 and K24..31.
__device__ __forceinline__ v16h load_fragB(const f16* tile, int row0, int col0, int ldt) {
  const int lane = threadIdx.x & 31;
  const f16* p = tile + (size_t)(row0 + (lane & 15)) * ldt + col0 + ((lane >> 4) << 3);
  v8h lo = *(const v8h*)p;
  v8h hi = *(const v8h*)(p + 16);
  return __builtin_shufflevector(lo, hi, 0,1,2,3,4,5,6,7,8,9,10,11,12,13,14,15);
}

__device__ __forceinline__ v8f wmma16(v16h a, v16h b, v8f c) {
  return __builtin_amdgcn_wmma_f32_16x16x32_f16(false, a, false, b, (short)0, c, false, false);
}

__device__ __forceinline__ float gelu_exact(float x) {
  return 0.5f * x * (1.0f + erff(x * 0.70710678118654752f));
}

// ---------------- weight convert + transpose: w[K][N] f32 -> wt[N][K] f16 ---

__global__ __launch_bounds__(256) void transpose_cvt_kernel(
    const float* __restrict__ w, f16* __restrict__ wt, int K, int N) {
  __shared__ float tile[32][33];
  const int kb = blockIdx.y * 32, nb = blockIdx.x * 32;
  const int tx = threadIdx.x & 31, ty = threadIdx.x >> 5;  // 32 x 8
  #pragma unroll
  for (int i = ty; i < 32; i += 8)
    tile[i][tx] = w[(size_t)(kb + i) * N + nb + tx];
  __syncthreads();
  #pragma unroll
  for (int i = ty; i < 32; i += 8)
    wt[(size_t)(nb + i) * K + kb + tx] = (f16)tile[tx][i];
}

// ---------------- LayerNorm (row = token, 1024 feats), f32 -> f16 -----------

__global__ __launch_bounds__(256) void ln_kernel(
    const float* __restrict__ x, const float* __restrict__ g,
    const float* __restrict__ b, f16* __restrict__ out) {
  const int row = blockIdx.x;
  const int tid = threadIdx.x;
  const float* xr = x + (size_t)row * FEATS;
  float s = 0.f, s2 = 0.f;
  #pragma unroll
  for (int i = tid; i < FEATS; i += 256) { float v = xr[i]; s += v; s2 += v * v; }
  __shared__ float rs[256], rq[256];
  rs[tid] = s; rq[tid] = s2;
  __syncthreads();
  for (int st = 128; st > 0; st >>= 1) {
    if (tid < st) { rs[tid] += rs[tid + st]; rq[tid] += rq[tid + st]; }
    __syncthreads();
  }
  const float mu  = rs[0] * (1.0f / FEATS);
  const float var = rq[0] * (1.0f / FEATS) - mu * mu;
  const float inv = rsqrtf(var + 1e-5f);
  f16* orow = out + (size_t)row * FEATS;
  #pragma unroll
  for (int i = tid; i < FEATS; i += 256)
    orow[i] = (f16)((xr[i] - mu) * inv * g[i] + b[i]);
}

// ---------------- WMMA GEMM: C = epi(A[M,K] @ BT[N,K]^T + bias) -------------
// 128x128 block, 8 waves (each 32x64), K staged 64 at a time with
// register-prefetch of the next global tile overlapping WMMA compute.

#define BM 128
#define BN 128
#define KC 64
#define LDT (KC + 8)   // padded LDS stride (f16), keeps 16B alignment

// epilogue modes
#define EPI_QK             0   // f16 store, layout [b,h,n,d]
#define EPI_V              1   // f16 store, layout [b,h,d,n] (pre-transposed V)
#define EPI_RESID_F32      2   // +resid, f32 store
#define EPI_GELU_F16       3   // gelu, f16 store
#define EPI_GELU_RESID_F32 4   // gelu, +resid, f32 store

__device__ __forceinline__ void gload_tile(const f16* __restrict__ base, int ldk,
                                           int brow, int k0, v8h* dst) {
  const int tid = threadIdx.x;
  #pragma unroll
  for (int it = 0; it < 4; ++it) {
    const int idx = tid + it * 256;          // 0..1023: 128 rows x 8 col-groups
    const int r = idx >> 3, cg = (idx & 7) << 3;
    dst[it] = *(const v8h*)&base[(size_t)(brow + r) * ldk + k0 + cg];
  }
}

__device__ __forceinline__ void sstore_tile(f16* sT, const v8h* src) {
  const int tid = threadIdx.x;
  #pragma unroll
  for (int it = 0; it < 4; ++it) {
    const int idx = tid + it * 256;
    const int r = idx >> 3, cg = (idx & 7) << 3;
    *(v8h*)&sT[r * LDT + cg] = src[it];
  }
}

template <int EPI>
__global__ __launch_bounds__(256) void gemm_kernel(
    const f16* __restrict__ A, const f16* __restrict__ BT,
    const float* __restrict__ bias, const float* __restrict__ resid,
    void* __restrict__ Cout, int M, int N, int K) {
  __shared__ f16 sA[BM * LDT];
  __shared__ f16 sB[BN * LDT];
  const int tid = threadIdx.x;
  const int lane = tid & 31, wave = tid >> 5;
  const int bm = blockIdx.y * BM, bn = blockIdx.x * BN;
  const int wm = (wave >> 1) * 32;  // 4 wave-rows of 32
  const int wn = (wave & 1) * 64;   // 2 wave-cols of 64

  v8f acc[2][4];
  #pragma unroll
  for (int i = 0; i < 2; ++i)
    #pragma unroll
    for (int j = 0; j < 4; ++j) acc[i][j] = (v8f)(0.0f);

  v8h pa[4], pb[4];
  gload_tile(A, K, bm, 0, pa);
  gload_tile(BT, K, bn, 0, pb);

  for (int k0 = 0; k0 < K; k0 += KC) {
    sstore_tile(sA, pa);
    sstore_tile(sB, pb);
    __syncthreads();
    if (k0 + KC < K) {                 // prefetch next stage during compute
      gload_tile(A, K, bm, k0 + KC, pa);
      gload_tile(BT, K, bn, k0 + KC, pb);
    }
    #pragma unroll
    for (int ks = 0; ks < 2; ++ks) {   // two 32-deep WMMA steps per stage
      v16h af[2], bf[4];
      #pragma unroll
      for (int mt = 0; mt < 2; ++mt) af[mt] = load_fragA(sA, wm + mt * 16, ks * 32, LDT);
      #pragma unroll
      for (int nt = 0; nt < 4; ++nt) bf[nt] = load_fragB(sB, wn + nt * 16, ks * 32, LDT);
      #pragma unroll
      for (int mt = 0; mt < 2; ++mt)
        #pragma unroll
        for (int nt = 0; nt < 4; ++nt)
          acc[mt][nt] = wmma16(af[mt], bf[nt], acc[mt][nt]);
    }
    __syncthreads();
  }

  // Epilogue. C/D layout: lanes 0-15 col N=lane, row = g; lanes 16-31 row = g+8.
  const int half8 = (lane >> 4) << 3;
  #pragma unroll
  for (int mt = 0; mt < 2; ++mt) {
    #pragma unroll
    for (int nt = 0; nt < 4; ++nt) {
      const int col = bn + wn + nt * 16 + (lane & 15);
      #pragma unroll
      for (int gg = 0; gg < 8; ++gg) {
        const int row = bm + wm + mt * 16 + gg + half8;
        float v = acc[mt][nt][gg] + bias[col];
        if (EPI == EPI_GELU_F16 || EPI == EPI_GELU_RESID_F32) v = gelu_exact(v);
        if (EPI == EPI_RESID_F32 || EPI == EPI_GELU_RESID_F32)
          v += resid[(size_t)row * N + col];
        if (EPI == EPI_QK) {
          const int b = row >> 10, n = row & (NSEQ - 1);
          const int h = col >> 6, dd = col & (DHEAD - 1);
          ((f16*)Cout)[(((size_t)(b * NHEAD + h) * NSEQ + n) << 6) + dd] = (f16)v;
        } else if (EPI == EPI_V) {
          const int b = row >> 10, n = row & (NSEQ - 1);
          const int h = col >> 6, dd = col & (DHEAD - 1);
          ((f16*)Cout)[((size_t)(b * NHEAD + h) * DHEAD + dd) * NSEQ + n] = (f16)v;
        } else if (EPI == EPI_GELU_F16) {
          ((f16*)Cout)[(size_t)row * N + col] = (f16)v;
        } else {
          ((float*)Cout)[(size_t)row * N + col] = v;
        }
      }
    }
  }
}

// ---------------- Flash attention per (b,h), 64 query rows per block --------
// q,k: [bh][n][64] f16 ; v: [bh][64][n] f16 ; out: [b][n][feats] f16
__global__ __launch_bounds__(128) void attn_kernel(
    const f16* __restrict__ q, const f16* __restrict__ k,
    const f16* __restrict__ v, f16* __restrict__ attn_out) {
  const int bh = blockIdx.y;          // 0..127
  const int r0 = blockIdx.x * 64;     // query-row tile
  const int tid = threadIdx.x, lane = tid & 31, wave = tid >> 5;
  const f16* qb = q + (size_t)bh * NSEQ * DHEAD;
  const f16* kb = k + (size_t)bh * NSEQ * DHEAD;
  const f16* vb = v + (size_t)bh * DHEAD * NSEQ;

  __shared__ f16 sQ[64 * 72];      // [row][d]
  __shared__ f16 sK[64 * 72];      // [key][d]
  __shared__ f16 sVt[64 * 72];     // [d][key]
  __shared__ f16 sP[4][16 * 72];   // per-wave P scratch [row][key]

  for (int idx = tid; idx < 64 * 8; idx += 128) {
    const int rr = idx >> 3, cg = (idx & 7) << 3;
    *(v8h*)&sQ[rr * 72 + cg] = *(const v8h*)&qb[(size_t)(r0 + rr) * DHEAD + cg];
  }
  __syncthreads();
  const v16h qf0 = load_fragA(sQ, wave * 16, 0, 72);
  const v16h qf1 = load_fragA(sQ, wave * 16, 32, 72);

  v8f oacc[4];
  #pragma unroll
  for (int nt = 0; nt < 4; ++nt) oacc[nt] = (v8f)(0.0f);
  float mstate[8], lstate[8];
  #pragma unroll
  for (int gg = 0; gg < 8; ++gg) { mstate[gg] = -1e30f; lstate[gg] = 0.0f; }

  const float sc = 1.0f / 32.0f;    // reference scales by sqrt(FEATS) = 32
  const int half8 = (lane >> 4) << 3;

  for (int kc = 0; kc < NSEQ; kc += 64) {
    for (int idx = tid; idx < 64 * 8; idx += 128) {
      const int rr = idx >> 3, cg = (idx & 7) << 3;
      *(v8h*)&sK[rr * 72 + cg]  = *(const v8h*)&kb[(size_t)(kc + rr) * DHEAD + cg];
      *(v8h*)&sVt[rr * 72 + cg] = *(const v8h*)&vb[(size_t)rr * NSEQ + kc + cg];
    }
    __syncthreads();

    // S = (Q @ K^T) / 32 : 4 key-tiles of 16, contraction d = 2x32
    v8f s[4];
    #pragma unroll
    for (int nt = 0; nt < 4; ++nt) {
      s[nt] = (v8f)(0.0f);
      s[nt] = wmma16(qf0, load_fragB(sK, nt * 16, 0, 72), s[nt]);
      s[nt] = wmma16(qf1, load_fragB(sK, nt * 16, 32, 72), s[nt]);
      #pragma unroll
      for (int gg = 0; gg < 8; ++gg) s[nt][gg] *= sc;
    }

    // online softmax: per-row (g, lane-half) reductions across 16 lanes
    float mnew[8], alpha[8];
    #pragma unroll
    for (int gg = 0; gg < 8; ++gg) {
      float mx = fmaxf(fmaxf(s[0][gg], s[1][gg]), fmaxf(s[2][gg], s[3][gg]));
      #pragma unroll
      for (int off = 1; off < 16; off <<= 1) mx = fmaxf(mx, __shfl_xor(mx, off, 32));
      mnew[gg] = fmaxf(mstate[gg], mx);
      alpha[gg] = __expf(mstate[gg] - mnew[gg]);
      mstate[gg] = mnew[gg];
    }
    #pragma unroll
    for (int gg = 0; gg < 8; ++gg) {
      float rsum = 0.0f;
      #pragma unroll
      for (int nt = 0; nt < 4; ++nt) {
        const float p = __expf(s[nt][gg] - mnew[gg]);
        s[nt][gg] = p; rsum += p;
      }
      #pragma unroll
      for (int off = 1; off < 16; off <<= 1) rsum += __shfl_xor(rsum, off, 32);
      lstate[gg] = lstate[gg] * alpha[gg] + rsum;
      #pragma unroll
      for (int nt = 0; nt < 4; ++nt) oacc[nt][gg] *= alpha[gg];
    }

    // C-layout P -> LDS -> A-layout fragments
    f16* myP = sP[wave];
    #pragma unroll
    for (int nt = 0; nt < 4; ++nt)
      #pragma unroll
      for (int gg = 0; gg < 8; ++gg)
        myP[(gg + half8) * 72 + nt * 16 + (lane & 15)] = (f16)s[nt][gg];
    const v16h pf0 = load_fragA(myP, 0, 0, 72);
    const v16h pf1 = load_fragA(myP, 0, 32, 72);

    // O += P @ V : contraction keys = 2x32, 4 d-tiles of 16
    #pragma unroll
    for (int nt = 0; nt < 4; ++nt) {
      oacc[nt] = wmma16(pf0, load_fragB(sVt, nt * 16, 0, 72), oacc[nt]);
      oacc[nt] = wmma16(pf1, load_fragB(sVt, nt * 16, 32, 72), oacc[nt]);
    }
    __syncthreads();
  }

  // normalize and store to attn[b][n][h*64+d]
  const int b = bh >> 4, h = bh & 15;
  #pragma unroll
  for (int nt = 0; nt < 4; ++nt) {
    const int col = h * DHEAD + nt * 16 + (lane & 15);
    #pragma unroll
    for (int gg = 0; gg < 8; ++gg) {
      const int rown = r0 + wave * 16 + gg + half8;
      attn_out[((size_t)b * NSEQ + rown) * FEATS + col] =
          (f16)(oacc[nt][gg] / lstate[gg]);
    }
  }
}

// ---------------------------------------------------------------------------

extern "C" void kernel_launch(void* const* d_in, const int* in_sizes, int n_in,
                              void* d_out, int out_size, void* d_ws, size_t ws_size,
                              hipStream_t stream) {
  const float* x    = (const float*)d_in[0];
  const float* ln1g = (const float*)d_in[1];
  const float* ln1b = (const float*)d_in[2];
  const float* wq   = (const float*)d_in[3];
  const float* bq   = (const float*)d_in[4];
  const float* wk   = (const float*)d_in[5];
  const float* bk   = (const float*)d_in[6];
  const float* wv   = (const float*)d_in[7];
  const float* bv   = (const float*)d_in[8];
  const float* wo   = (const float*)d_in[9];
  const float* bo   = (const float*)d_in[10];
  const float* ln2g = (const float*)d_in[11];
  const float* ln2b = (const float*)d_in[12];
  const float* w1   = (const float*)d_in[13];
  const float* b1   = (const float*)d_in[14];
  const float* w2   = (const float*)d_in[15];
  const float* b2   = (const float*)d_in[16];
  float* out = (float*)d_out;

  char* ws = (char*)d_ws;
  const size_t MB = 1024ull * 1024ull;
  f16* wqT  = (f16*)(ws + 0 * MB);    // [1024][1024]
  f16* wkT  = (f16*)(ws + 2 * MB);
  f16* wvT  = (f16*)(ws + 4 * MB);
  f16* woT  = (f16*)(ws + 6 * MB);
  f16* w1T  = (f16*)(ws + 8 * MB);    // [4096][1024]
  f16* w2T  = (f16*)(ws + 16 * MB);   // [1024][4096]
  f16* h    = (f16*)(ws + 24 * MB);   // ln output, reused for ln2 output
  f16* qb   = (f16*)(ws + 40 * MB);   // [b,h,n,d]
  f16* kb   = (f16*)(ws + 56 * MB);   // [b,h,n,d]
  f16* vb   = (f16*)(ws + 72 * MB);   // [b,h,d,n]
  f16* attn = (f16*)(ws + 88 * MB);   // [b,n,f]
  f16* mbuf = (f16*)(ws + 40 * MB);   // reuses q/k/v/attn region after attention
  float* res = (float*)(ws + 104 * MB); // residual1 (f32), 32MB -> 136MB total

  const dim3 tb(256);
  // weight transpose+convert
  transpose_cvt_kernel<<<dim3(FEATS / 32, FEATS / 32), tb, 0, stream>>>(wq, wqT, FEATS, FEATS);
  transpose_cvt_kernel<<<dim3(FEATS / 32, FEATS / 32), tb, 0, stream>>>(wk, wkT, FEATS, FEATS);
  transpose_cvt_kernel<<<dim3(FEATS / 32, FEATS / 32), tb, 0, stream>>>(wv, wvT, FEATS, FEATS);
  transpose_cvt_kernel<<<dim3(FEATS / 32, FEATS / 32), tb, 0, stream>>>(wo, woT, FEATS, FEATS);
  transpose_cvt_kernel<<<dim3(MLPH / 32, FEATS / 32), tb, 0, stream>>>(w1, w1T, FEATS, MLPH);
  transpose_cvt_kernel<<<dim3(FEATS / 32, MLPH / 32), tb, 0, stream>>>(w2, w2T, MLPH, FEATS);

  // LN1
  ln_kernel<<<NTOK, tb, 0, stream>>>(x, ln1g, ln1b, h);

  // QKV projections (Q/K standard layout, V pre-transposed)
  const dim3 gqkv(FEATS / BN, NTOK / BM);
  gemm_kernel<EPI_QK><<<gqkv, tb, 0, stream>>>(h, wqT, bq, nullptr, qb, NTOK, FEATS, FEATS);
  gemm_kernel<EPI_QK><<<gqkv, tb, 0, stream>>>(h, wkT, bk, nullptr, kb, NTOK, FEATS, FEATS);
  gemm_kernel<EPI_V ><<<gqkv, tb, 0, stream>>>(h, wvT, bv, nullptr, vb, NTOK, FEATS, FEATS);

  // attention
  attn_kernel<<<dim3(NSEQ / 64, NBATCH * NHEAD), dim3(128), 0, stream>>>(qb, kb, vb, attn);

  // O projection + residual (x)
  gemm_kernel<EPI_RESID_F32><<<gqkv, tb, 0, stream>>>(attn, woT, bo, x, res, NTOK, FEATS, FEATS);

  // LN2
  ln_kernel<<<NTOK, tb, 0, stream>>>(res, ln2g, ln2b, h);

  // MLP1: gelu(h @ w1 + b1) -> f16
  gemm_kernel<EPI_GELU_F16><<<dim3(MLPH / BN, NTOK / BM), tb, 0, stream>>>(
      h, w1T, b1, nullptr, mbuf, NTOK, MLPH, FEATS);

  // MLP2: gelu(m @ w2 + b2) + res -> f32 output
  gemm_kernel<EPI_GELU_RESID_F32><<<dim3(FEATS / BN, NTOK / BM), tb, 0, stream>>>(
      mbuf, w2T, b2, res, out, NTOK, FEATS, MLPH);
}
